// GroupQueryAttention_31928786878514
// MI455X (gfx1250) — matrix operand order repaired and checked by reading, not compiled, not run on hardware
//
#include <hip/hip_runtime.h>
#include <hip/hip_bf16.h>
#include <stdint.h>
#include <stddef.h>

// ---------------------------------------------------------------------------
// GQA attention layer, bf16 WMMA pipeline for gfx1250 (MI455X).
//   dims: HIDDEN=2048, NH=32, NKV=8, HD=64, GROUP=4, S=2048, B derived.
// This round: double-buffered async global->LDS staging (ASYNCcnt) so the
// DMA of tile i+1 overlaps the WMMAs of tile i; one barrier per K-step.
// ---------------------------------------------------------------------------

#define HIDDEN 2048
#define NH     32
#define NKV    8
#define HD     64
#define SEQ    2048

typedef __attribute__((ext_vector_type(16))) __bf16 v16bf;
typedef __attribute__((ext_vector_type(8)))  __bf16 v8bf;
typedef __attribute__((ext_vector_type(4)))  __bf16 v4bf;
typedef __attribute__((ext_vector_type(8)))  float  v8f;

union Frag16 { v16bf v; v8bf h[2]; };

__device__ __forceinline__ __bf16 f2bf(float f) {
    unsigned u = __builtin_bit_cast(unsigned, f);
    unsigned r = u + 0x7FFFu + ((u >> 16) & 1u);   // round-to-nearest-even
    unsigned short hs = (unsigned short)(r >> 16);
    return __builtin_bit_cast(__bf16, hs);
}

__device__ __forceinline__ v8f wmma_bf16(v16bf a, v16bf b, v8f c) {
    return __builtin_amdgcn_wmma_f32_16x16x32_bf16(
        /*neg_a=*/false, a, /*neg_b=*/false, b,
        /*c_mod=*/(short)0, c, /*reuse_a=*/false, /*reuse_b=*/false);
}

// Async global -> LDS copy of 16 bytes per lane (GLOBAL_LOAD_ASYNC_TO_LDS_B128).
// LDS offset = low 32 bits of the generic LDS address.
__device__ __forceinline__ void async_copy_b128(void* lds_dst, const void* gsrc) {
    unsigned lds_off = (unsigned)(uintptr_t)lds_dst;
    unsigned long long ga = (unsigned long long)(uintptr_t)gsrc;
    asm volatile("global_load_async_to_lds_b128 %0, %1, off"
                 :: "v"(lds_off), "v"(ga) : "memory");
}

__device__ __forceinline__ void wait_asynccnt0() {
    asm volatile("s_wait_asynccnt 0x0" ::: "memory");
}

// Compiler-only ordering fence: DS ops are in-order per wave (ISA 7.3), so
// intra-wave LDS RAW needs no barrier, only source-order preservation.
__device__ __forceinline__ void compiler_fence() {
    asm volatile("" ::: "memory");
}

// ---------------------------------------------------------------------------
// fp32 -> bf16 conversion (vectorized by 4)
// ---------------------------------------------------------------------------
__global__ void cvt_f32_to_bf16(const float* __restrict__ in,
                                __bf16* __restrict__ out, int n4) {
    int i = blockIdx.x * blockDim.x + threadIdx.x;
    if (i < n4) {
        float4 f = ((const float4*)in)[i];
        v4bf o;
        o[0] = f2bf(f.x); o[1] = f2bf(f.y); o[2] = f2bf(f.z); o[3] = f2bf(f.w);
        ((v4bf*)out)[i] = o;
    }
}

// ---------------------------------------------------------------------------
// Tiled bf16 GEMM + bias.  C[M,N] = A[M,K] @ W[K,N] + bias.
// 256 threads = 8 waves; block tile 128(M) x 64(N); K step 32, double-buffered.
// MODE 0: fp32 row-major output (final projection).
// MODE 1: bf16 head-major output [B, heads, S, 64] (QKV projections).
// ---------------------------------------------------------------------------
template <int MODE>
__global__ __launch_bounds__(256)
void gemm_bias(const __bf16* __restrict__ A, const __bf16* __restrict__ W,
               const float* __restrict__ bias, float* __restrict__ outF,
               __bf16* __restrict__ outB, int M, int N, int K, int S, int heads) {
    __shared__ __align__(16) __bf16 lsA[2][128][32];
    __shared__ __align__(16) __bf16 lsWt[2][64][32];   // transposed: [n][k]

    const int tid  = threadIdx.x;
    const int lane = tid & 31;
    const int wave = tid >> 5;
    const int wm   = wave >> 1;        // 0..3
    const int wn   = wave & 1;         // 0..1
    const int mtile = blockIdx.y * 128;
    const int ntile = blockIdx.x * 64;

    const int l15  = lane & 15;
    const int kb8  = (lane >> 4) * 8;   // A-frag K base (16-bit A layout)
    const int kb16 = (lane >> 4) * 16;  // unused; B frags also use kb8

    const int rowA  = tid >> 1;         // 0..127
    const int halfA = tid & 1;          // 0..1 (16 elems each)
    const int kW    = tid >> 3;         // 0..31
    const int n0W   = (tid & 7) * 8;    // 0..56

    const __bf16* Arow = A + (size_t)(mtile + rowA) * K + halfA * 16;
    const __bf16* Wrow = W + (size_t)kW * N + ntile + n0W;

    const v8f vzero = {0.f,0.f,0.f,0.f,0.f,0.f,0.f,0.f};
    v8f acc[2][2];
    acc[0][0] = vzero; acc[0][1] = vzero; acc[1][0] = vzero; acc[1][1] = vzero;

    // Prologue: stage tile 0 into buffer 0.
    async_copy_b128(&lsA[0][rowA][halfA * 16], Arow);
    {
        v8bf w0 = *(const v8bf*)Wrow;
        #pragma unroll
        for (int i = 0; i < 8; ++i) lsWt[0][n0W + i][kW] = w0[i];
    }
    wait_asynccnt0();
    __syncthreads();

    int p = 0;
    for (int kb = 0; kb < K; kb += 32) {
        const bool has_next = (kb + 32 < K);
        v8bf wn_next = {};
        if (has_next) {
            // Kick off next tile's DMA + W prefetch; overlaps with WMMAs below.
            async_copy_b128(&lsA[1 - p][rowA][halfA * 16], Arow + kb + 32);
            wn_next = *(const v8bf*)(Wrow + (size_t)(kb + 32) * N);
        }

        Frag16 af[2], bfr[2];
        #pragma unroll
        for (int mi = 0; mi < 2; ++mi) {
            const int r = wm * 32 + mi * 16 + l15;
            af[mi].h[0] = *(const v8bf*)&lsA[p][r][kb8];
            af[mi].h[1] = *(const v8bf*)&lsA[p][r][kb8 + 16];
        }
        #pragma unroll
        for (int ni = 0; ni < 2; ++ni) {
            const int n = wn * 32 + ni * 16 + l15;
            bfr[ni].h[0] = *(const v8bf*)&lsWt[p][n][kb8];
            bfr[ni].h[1] = *(const v8bf*)&lsWt[p][n][kb8 + 16];
        }
        #pragma unroll
        for (int mi = 0; mi < 2; ++mi)
            #pragma unroll
            for (int ni = 0; ni < 2; ++ni)
                acc[mi][ni] = wmma_bf16(af[mi].v, bfr[ni].v, acc[mi][ni]);

        if (has_next) {
            #pragma unroll
            for (int i = 0; i < 8; ++i) lsWt[1 - p][n0W + i][kW] = wn_next[i];
        }
        wait_asynccnt0();
        __syncthreads();
        p ^= 1;
    }

    // Epilogue. C/D layout: reg j, lanes 0-15 -> row j, lanes 16-31 -> row j+8.
    #pragma unroll
    for (int mi = 0; mi < 2; ++mi) {
        #pragma unroll
        for (int ni = 0; ni < 2; ++ni) {
            #pragma unroll
            for (int j = 0; j < 8; ++j) {
                const int m = mtile + wm * 32 + mi * 16 + j + ((lane >> 4) << 3);
                const int n = ntile + wn * 32 + ni * 16 + l15;
                const float val = acc[mi][ni][j] + bias[n];
                if (MODE == 0) {
                    outF[(size_t)m * N + n] = val;
                } else {
                    const int b = m / S, s = m % S;
                    const int h = n >> 6, d = n & 63;
                    outB[(((size_t)b * heads + h) * S + s) * HD + d] = f2bf(val);
                }
            }
        }
    }
}

// ---------------------------------------------------------------------------
// Flash-style GQA attention. 128 threads = 4 waves per (b, h, 64-query block).
// Key blocks of 64 double-buffered in LDS: K via async DMA, V via VGPR
// transpose overlapped with the score WMMAs. One barrier per key block.
// ---------------------------------------------------------------------------
__global__ __launch_bounds__(128)
void gqa_attention(const __bf16* __restrict__ Q, const __bf16* __restrict__ Kmat,
                   const __bf16* __restrict__ Vmat, __bf16* __restrict__ out,
                   int S, int B) {
    __shared__ __align__(16) __bf16 lsK[2][64][64];    // [key][d]  (natural)
    __shared__ __align__(16) __bf16 lsVt[2][64][64];   // [d][key]  (transposed)
    __shared__ __align__(16) __bf16 lsP[4][16][64];    // per-wave P tile

    const int tid  = threadIdx.x;
    const int lane = tid & 31;
    const int wave = tid >> 5;
    const int b  = blockIdx.z;
    const int h  = blockIdx.y;
    const int qb = blockIdx.x * 64;
    const int kvh = h >> 2;                            // GROUP = 4

    const __bf16* Qp = Q    + (((size_t)b * NH  + h)   * S + qb + wave * 16) * HD;
    const __bf16* Kp = Kmat + (((size_t)b * NKV + kvh) * S) * HD;
    const __bf16* Vp = Vmat + (((size_t)b * NKV + kvh) * S) * HD;

    const int l15  = lane & 15;
    const int kb8  = (lane >> 4) * 8;
    const int kb16 = (lane >> 4) * 16;

    // Q fragments (2 k-chunks of 32 over hd=64), loaded once.
    Frag16 qf[2];
    #pragma unroll
    for (int c = 0; c < 2; ++c) {
        qf[c].h[0] = *(const v8bf*)(Qp + (size_t)l15 * HD + c * 32 + kb8);
        qf[c].h[1] = *(const v8bf*)(Qp + (size_t)l15 * HD + c * 32 + kb8 + 16);
    }

    const v8f vzero = {0.f,0.f,0.f,0.f,0.f,0.f,0.f,0.f};
    v8f acc_o[4];
    #pragma unroll
    for (int t = 0; t < 4; ++t) acc_o[t] = vzero;
    float run_m[8], run_l[8];
    #pragma unroll
    for (int j = 0; j < 8; ++j) { run_m[j] = -1.0e30f; run_l[j] = 0.f; }

    const float scale = 0.125f;                        // 1/sqrt(64)
    const int keyS = tid & 63;                         // staging: key row
    const int keyH = (tid >> 6) * 32;                  // staging: d half

    // Prologue: stage key block 0 into buffer 0.
    #pragma unroll
    for (int i = 0; i < 4; ++i) {
        async_copy_b128(&lsK[0][keyS][keyH + i * 8],
                        Kp + (size_t)keyS * HD + keyH + i * 8);
        v8bf v = *(const v8bf*)(Vp + (size_t)keyS * HD + keyH + i * 8);
        #pragma unroll
        for (int e = 0; e < 8; ++e) lsVt[0][keyH + i * 8 + e][keyS] = v[e];
    }
    wait_asynccnt0();
    __syncthreads();

    int p = 0;
    for (int kb = 0; kb < S; kb += 64) {
        const bool has_next = (kb + 64 < S);
        v8bf vv[4] = {};
        if (has_next) {
            #pragma unroll
            for (int i = 0; i < 4; ++i) {
                async_copy_b128(&lsK[1 - p][keyS][keyH + i * 8],
                                Kp + (size_t)(kb + 64 + keyS) * HD + keyH + i * 8);
                vv[i] = *(const v8bf*)(Vp + (size_t)(kb + 64 + keyS) * HD + keyH + i * 8);
            }
        }

        // Scores: 16 q-rows x 64 keys = 4 N-tiles x 2 k-chunks of WMMA.
        v8f sc[4];
        #pragma unroll
        for (int t = 0; t < 4; ++t) sc[t] = vzero;
        #pragma unroll
        for (int t = 0; t < 4; ++t) {
            #pragma unroll
            for (int c = 0; c < 2; ++c) {
                Frag16 kf;
                kf.h[0] = *(const v8bf*)&lsK[p][t * 16 + l15][c * 32 + kb8];
                kf.h[1] = *(const v8bf*)&lsK[p][t * 16 + l15][c * 32 + kb8 + 16];
                sc[t] = wmma_bf16(qf[c].v, kf.v, sc[t]);
            }
        }

        // Online softmax per row (rows live across 16-lane half-waves).
        #pragma unroll
        for (int j = 0; j < 8; ++j) {
            float mx = sc[0][j];
            #pragma unroll
            for (int t = 1; t < 4; ++t) mx = fmaxf(mx, sc[t][j]);
            #pragma unroll
            for (int off = 1; off < 16; off <<= 1)
                mx = fmaxf(mx, __shfl_xor(mx, off, 32));
            const float nm   = fmaxf(run_m[j], mx * scale);
            const float corr = __expf(run_m[j] - nm);
            float rs = 0.f;
            const int prow = j + ((lane >> 4) << 3);
            #pragma unroll
            for (int t = 0; t < 4; ++t) {
                const float p2 = __expf(sc[t][j] * scale - nm);
                rs += p2;
                lsP[wave][prow][t * 16 + l15] = f2bf(p2);
            }
            #pragma unroll
            for (int off = 1; off < 16; off <<= 1)
                rs += __shfl_xor(rs, off, 32);
            run_l[j] = run_l[j] * corr + rs;
            run_m[j] = nm;
            #pragma unroll
            for (int t = 0; t < 4; ++t) acc_o[t][j] *= corr;
        }
        // Intra-wave LDS RAW on lsP: DS pipe is in-order per wave; just keep
        // source order (no block barrier needed).
        compiler_fence();

        // Scatter next V block into the alternate transposed buffer.
        if (has_next) {
            #pragma unroll
            for (int i = 0; i < 4; ++i)
                #pragma unroll
                for (int e = 0; e < 8; ++e)
                    lsVt[1 - p][keyH + i * 8 + e][keyS] = vv[i][e];
        }

        // PV: acc_o += P(16x64) @ V(64x64), 2 k-chunks x 4 N-tiles.
        #pragma unroll
        for (int c = 0; c < 2; ++c) {
            Frag16 pf;
            pf.h[0] = *(const v8bf*)&lsP[wave][l15][c * 32 + kb8];
            pf.h[1] = *(const v8bf*)&lsP[wave][l15][c * 32 + kb8 + 16];
            #pragma unroll
            for (int t = 0; t < 4; ++t) {
                Frag16 vf;
                vf.h[0] = *(const v8bf*)&lsVt[p][t * 16 + l15][c * 32 + kb8];
                vf.h[1] = *(const v8bf*)&lsVt[p][t * 16 + l15][c * 32 + kb8 + 16];
                acc_o[t] = wmma_bf16(pf.v, vf.v, acc_o[t]);
            }
        }

        wait_asynccnt0();
        __syncthreads();
        p ^= 1;
    }

    // Normalize and write attention output directly in [B, S, 2048] bf16.
    #pragma unroll
    for (int j = 0; j < 8; ++j) {
        const float inv = 1.0f / run_l[j];
        const int s = qb + wave * 16 + j + ((lane >> 4) << 3);
        #pragma unroll
        for (int t = 0; t < 4; ++t) {
            const int d = t * 16 + l15;
            out[((size_t)b * S + s) * HIDDEN + h * HD + d] =
                f2bf(acc_o[t][j] * inv);
        }
    }
}

// ---------------------------------------------------------------------------
// Host-side orchestration
// ---------------------------------------------------------------------------
extern "C" void kernel_launch(void* const* d_in, const int* in_sizes, int n_in,
                              void* d_out, int out_size, void* d_ws, size_t ws_size,
                              hipStream_t stream) {
    const float* x  = (const float*)d_in[0];
    const float* Wq = (const float*)d_in[1];
    const float* bq = (const float*)d_in[2];
    const float* Wk = (const float*)d_in[3];
    const float* bk = (const float*)d_in[4];
    const float* Wv = (const float*)d_in[5];
    const float* bv = (const float*)d_in[6];
    const float* Wo = (const float*)d_in[7];
    const float* bo = (const float*)d_in[8];

    const int B = in_sizes[0] / (SEQ * HIDDEN);   // = 2
    const int M = B * SEQ;                        // = 4096
    const int NKVD = NKV * HD;                    // = 512

    // Workspace carve-up (bf16 buffers)
    char* w = (char*)d_ws;
    __bf16* xb   = (__bf16*)w;  w += (size_t)M * HIDDEN * 2;
    __bf16* wqb  = (__bf16*)w;  w += (size_t)HIDDEN * HIDDEN * 2;
    __bf16* wkb  = (__bf16*)w;  w += (size_t)HIDDEN * NKVD * 2;
    __bf16* wvb  = (__bf16*)w;  w += (size_t)HIDDEN * NKVD * 2;
    __bf16* wob  = (__bf16*)w;  w += (size_t)HIDDEN * HIDDEN * 2;
    __bf16* qbuf = (__bf16*)w;  w += (size_t)M * HIDDEN * 2;   // [B,NH,S,64]
    __bf16* kbuf = (__bf16*)w;  w += (size_t)M * NKVD * 2;     // [B,NKV,S,64]
    __bf16* vbuf = (__bf16*)w;  w += (size_t)M * NKVD * 2;
    __bf16* xat  = (__bf16*)w;  w += (size_t)M * HIDDEN * 2;   // [B,S,2048]

    auto cvt = [&](const float* src, __bf16* dst, size_t n) {
        int n4 = (int)(n / 4);
        cvt_f32_to_bf16<<<(n4 + 255) / 256, 256, 0, stream>>>(src, dst, n4);
    };
    cvt(x,  xb,  (size_t)M * HIDDEN);
    cvt(Wq, wqb, (size_t)HIDDEN * HIDDEN);
    cvt(Wk, wkb, (size_t)HIDDEN * NKVD);
    cvt(Wv, wvb, (size_t)HIDDEN * NKVD);
    cvt(Wo, wob, (size_t)HIDDEN * HIDDEN);

    // Q/K/V projections -> head-major bf16
    gemm_bias<1><<<dim3(HIDDEN / 64, M / 128), 256, 0, stream>>>(
        xb, wqb, bq, nullptr, qbuf, M, HIDDEN, HIDDEN, SEQ, NH);
    gemm_bias<1><<<dim3(NKVD / 64, M / 128), 256, 0, stream>>>(
        xb, wkb, bk, nullptr, kbuf, M, NKVD, HIDDEN, SEQ, NKV);
    gemm_bias<1><<<dim3(NKVD / 64, M / 128), 256, 0, stream>>>(
        xb, wvb, bv, nullptr, vbuf, M, NKVD, HIDDEN, SEQ, NKV);

    // Attention
    gqa_attention<<<dim3(SEQ / 64, NH, B), 128, 0, stream>>>(
        qbuf, kbuf, vbuf, xat, SEQ, B);

    // Output projection -> fp32
    gemm_bias<0><<<dim3(HIDDEN / 64, M / 128), 256, 0, stream>>>(
        xat, wob, bo, (float*)d_out, nullptr, M, HIDDEN, HIDDEN, SEQ, NH);

    (void)n_in; (void)out_size; (void)ws_size;
}
